// CausalSelfAttention_14637248545157
// MI455X (gfx1250) — matrix operand order repaired and checked by reading, not compiled, not run on hardware
//
#include <hip/hip_runtime.h>

// ---------------------------------------------------------------------------
// CDNA5 (gfx1250) causal self-attention, bf16 WMMA path.
// B=2, T=2048, C=1024, H=16, HS=64
// ---------------------------------------------------------------------------

typedef __attribute__((ext_vector_type(16))) __bf16 v16bf;
typedef __attribute__((ext_vector_type(8)))  float  v8f;

union F8   { v8f   v; float        f[8]; };
union AB16 { v16bf v; unsigned int u[8]; };

__device__ __forceinline__ __bf16 f2bf(float x) {
  unsigned int u = __builtin_bit_cast(unsigned int, x);
  unsigned int rnd = u + 0x7FFFu + ((u >> 16) & 1u);   // round-to-nearest-even
  unsigned short h = (unsigned short)(rnd >> 16);
  return __builtin_bit_cast(__bf16, h);
}

// A fragment: 16x32 (MxK) bf16, row-major tile in LDS, row stride `ld`.
// Per ISA 7.12.2: lane r=lane&15 holds row M=r; VGPR v holds K pair at
// k = 2v + (v>=4 ? 8 : 0) + 8*(lane>>4).
__device__ __forceinline__ v16bf load_a_frag(const __bf16* base, int ld) {
  const int lane = threadIdx.x & 31;
  const int r = lane & 15;
  const int hl = lane >> 4;
  AB16 f;
  const __bf16* row = base + r * ld;
#pragma unroll
  for (int v = 0; v < 8; ++v) {
    const int k = 2 * v + ((v >= 4) ? 8 : 0) + 8 * hl;
    f.u[v] = *reinterpret_cast<const unsigned int*>(row + k);
  }
  return f.v;
}

// B fragment: 32x16 (KxN) bf16, supplied as a TRANSPOSED tile Bt[n][k] in LDS
// with row stride `ld`. Lane r=lane&15 holds column N=r; VGPR v holds K pair
// at k = 2v + (v>=4 ? 8 : 0) + 8*(lane>>4).
__device__ __forceinline__ v16bf load_bt_frag(const __bf16* base, int ld) {
  const int lane = threadIdx.x & 31;
  const int r = lane & 15;
  const int hl = lane >> 4;
  AB16 f;
  const __bf16* row = base + r * ld;
#pragma unroll
  for (int v = 0; v < 8; ++v) {
    const int k = 2 * v + ((v >= 4) ? 8 : 0) + 8 * hl;
    f.u[v] = *reinterpret_cast<const unsigned int*>(row + k);
  }
  return f.v;
}

#define WMMA_BF16(a, b, c) \
  __builtin_amdgcn_wmma_f32_16x16x32_bf16(false, (a), false, (b), (short)0, (c), false, false)

// ---------------------------------------------------------------------------
// Kernel 1: qkv = x @ w_attn + b_attn  (M=4096, N=3072, K=1024)
// Converts fp32->bf16 while staging. Scatters Q,K as [B*H][T][64] and
// V transposed as [B*H][64][T] (bf16).
// ---------------------------------------------------------------------------
__global__ __launch_bounds__(256) void qkv_gemm_kernel(
    const float* __restrict__ x, const float* __restrict__ w,
    const float* __restrict__ bias,
    __bf16* __restrict__ Qo, __bf16* __restrict__ Ko, __bf16* __restrict__ Vto)
{
  __shared__ __attribute__((aligned(16))) __bf16 ldsA[128 * 64];   // A[m][k]
  __shared__ __attribute__((aligned(16))) __bf16 ldsB[128 * 64];   // Bt[n][k]
  const int tid  = threadIdx.x;
  const int m0   = blockIdx.y * 128;
  const int n0   = blockIdx.x * 128;
  const int wv   = tid >> 5;
  const int lane = tid & 31;
  const int r    = lane & 15;
  const int hl   = lane >> 4;

  F8 acc[8];
#pragma unroll
  for (int s = 0; s < 8; ++s)
#pragma unroll
    for (int v = 0; v < 8; ++v) acc[s].f[v] = 0.0f;

  for (int k0 = 0; k0 < 1024; k0 += 64) {
    __syncthreads();
    // Stage A tile 128x64: x fp32 -> bf16
#pragma unroll
    for (int i = 0; i < 8; ++i) {
      const int e   = (i * 256 + tid) * 4;
      const int row = e >> 6, col = e & 63;
      const float4 xa =
          *reinterpret_cast<const float4*>(x + (size_t)(m0 + row) * 1024 + k0 + col);
      __bf16* dst = &ldsA[row * 64 + col];
      dst[0] = f2bf(xa.x); dst[1] = f2bf(xa.y); dst[2] = f2bf(xa.z); dst[3] = f2bf(xa.w);
    }
    // Stage Bt tile: w[k][n] fp32 -> bf16 transposed into Bt[n][k]
#pragma unroll
    for (int i = 0; i < 8; ++i) {
      const int e  = (i * 256 + tid) * 4;
      const int kk = e >> 7, nn = e & 127;
      const float4 wa =
          *reinterpret_cast<const float4*>(w + (size_t)(k0 + kk) * 3072 + n0 + nn);
      ldsB[(nn + 0) * 64 + kk] = f2bf(wa.x);
      ldsB[(nn + 1) * 64 + kk] = f2bf(wa.y);
      ldsB[(nn + 2) * 64 + kk] = f2bf(wa.z);
      ldsB[(nn + 3) * 64 + kk] = f2bf(wa.w);
    }
    __syncthreads();
#pragma unroll
    for (int kk = 0; kk < 64; kk += 32) {
      const v16bf a = load_a_frag(&ldsA[(wv * 16) * 64 + kk], 64);
#pragma unroll
      for (int s = 0; s < 8; ++s) {
        const v16bf b = load_bt_frag(&ldsB[(s * 16) * 64 + kk], 64);
        acc[s].v = WMMA_BF16(a, b, acc[s].v);
      }
    }
  }

  // Epilogue: bias add + scatter to Q / K / V^T (bf16)
#pragma unroll
  for (int s = 0; s < 8; ++s) {
    const int n    = n0 + s * 16 + r;
    const float bn = bias[n];
    const int part = n >> 10;          // 0=q, 1=k, 2=v
    const int c    = n & 1023;
    const int hh   = c >> 6, d = c & 63;
#pragma unroll
    for (int v = 0; v < 8; ++v) {
      const float val = acc[s].f[v] + bn;
      const int m  = m0 + wv * 16 + v + 8 * hl;   // flat token index in [0, B*T)
      const int b  = m >> 11, t = m & 2047;
      const int bh = b * 16 + hh;
      if (part == 0)      Qo [((size_t)bh * 2048 + t) * 64 + d] = f2bf(val);
      else if (part == 1) Ko [((size_t)bh * 2048 + t) * 64 + d] = f2bf(val);
      else                Vto[((size_t)bh * 64 + d) * 2048 + t] = f2bf(val);
    }
  }
}

// ---------------------------------------------------------------------------
// Kernel 2: flash attention. Block = (b*h, 64-row q tile), 4 waves x 16 rows.
// Online softmax, 32 keys per step, all matmuls via WMMA bf16.
// ---------------------------------------------------------------------------
__global__ __launch_bounds__(128) void flash_attn_kernel(
    const __bf16* __restrict__ Q, const __bf16* __restrict__ K,
    const __bf16* __restrict__ Vt, __bf16* __restrict__ Y)
{
  __shared__ __attribute__((aligned(16))) __bf16 ldsQ[64 * 64];     // [q][d]
  __shared__ __attribute__((aligned(16))) __bf16 ldsK[32 * 64];     // [key][d] == Bt for QK^T
  __shared__ __attribute__((aligned(16))) __bf16 ldsV[64 * 32];     // [d][key] == Bt for P@V
  __shared__ __attribute__((aligned(16))) __bf16 ldsP[4][16 * 32];  // per-wave P

  const int tid  = threadIdx.x;
  const int wv   = tid >> 5;
  const int lane = tid & 31;
  const int r    = lane & 15;
  const int hl   = lane >> 4;
  const int bh   = blockIdx.x;
  const int tile = blockIdx.y;
  const int q0   = tile * 64 + wv * 16;

  // Stage Q tile [64][64]
  {
    const __bf16* src = Q + ((size_t)bh * 2048 + tile * 64) * 64;
#pragma unroll
    for (int i = 0; i < 8; ++i) {
      const int idx = i * 128 + tid;                // 1024 uint2
      reinterpret_cast<uint2*>(ldsQ)[idx] = reinterpret_cast<const uint2*>(src)[idx];
    }
  }
  __syncthreads();

  const v16bf qa0 = load_a_frag(&ldsQ[(wv * 16) * 64 + 0], 64);
  const v16bf qa1 = load_a_frag(&ldsQ[(wv * 16) * 64 + 32], 64);

  F8 acc[4];
#pragma unroll
  for (int s = 0; s < 4; ++s)
#pragma unroll
    for (int v = 0; v < 8; ++v) acc[s].f[v] = 0.0f;
  float m_i[8], l_i[8];
#pragma unroll
  for (int v = 0; v < 8; ++v) { m_i[v] = -__builtin_inff(); l_i[v] = 0.0f; }

  const int kt_end = tile * 64 + 64;
  for (int kt = 0; kt < kt_end; kt += 32) {
    __syncthreads();
    // Stage K tile [32 keys][64 d]
    {
      const __bf16* src = K + ((size_t)bh * 2048 + kt) * 64;
#pragma unroll
      for (int i = 0; i < 4; ++i) {
        const int idx = i * 128 + tid;              // 512 uint2
        reinterpret_cast<uint2*>(ldsK)[idx] = reinterpret_cast<const uint2*>(src)[idx];
      }
    }
    // Stage V tile [64 d][32 keys] from V^T
    {
#pragma unroll
      for (int i = 0; i < 2; ++i) {
        const int idx = i * 128 + tid;              // 256 uint4
        const int d   = idx >> 2;
        const int col = (idx & 3) * 8;
        const __bf16* src = Vt + ((size_t)bh * 64 + d) * 2048 + kt + col;
        *reinterpret_cast<uint4*>(&ldsV[d * 32 + col]) =
            *reinterpret_cast<const uint4*>(src);
      }
    }
    __syncthreads();

    if (kt <= q0 + 15) {   // wave-uniform: skip fully-masked key steps
      // S[16 q][32 keys] = Q @ K^T
      F8 s0, s1;
      {
        v8f t0 = {0,0,0,0,0,0,0,0};
        v8f t1 = {0,0,0,0,0,0,0,0};
        const v16bf kb00 = load_bt_frag(&ldsK[ 0 * 64 +  0], 64);
        const v16bf kb01 = load_bt_frag(&ldsK[ 0 * 64 + 32], 64);
        t0 = WMMA_BF16(qa0, kb00, t0);
        t0 = WMMA_BF16(qa1, kb01, t0);
        const v16bf kb10 = load_bt_frag(&ldsK[16 * 64 +  0], 64);
        const v16bf kb11 = load_bt_frag(&ldsK[16 * 64 + 32], 64);
        t1 = WMMA_BF16(qa0, kb10, t1);
        t1 = WMMA_BF16(qa1, kb11, t1);
        s0.v = t0; s1.v = t1;
      }
      const int kg0 = kt + r, kg1 = kt + 16 + r;
#pragma unroll
      for (int v = 0; v < 8; ++v) {
        const int qg = q0 + v + 8 * hl;
        float e0 = s0.f[v] * 0.125f;                 // 1/sqrt(64)
        float e1 = s1.f[v] * 0.125f;
        if (kg0 > qg) e0 = -__builtin_inff();
        if (kg1 > qg) e1 = -__builtin_inff();
        float rmax = fmaxf(e0, e1);
#pragma unroll
        for (int msk = 1; msk < 16; msk <<= 1)
          rmax = fmaxf(rmax, __shfl_xor(rmax, msk, 32));   // reduce within lane half
        const float mn   = fmaxf(m_i[v], rmax);
        const float corr = __expf(m_i[v] - mn);
        const float p0   = __expf(e0 - mn);
        const float p1   = __expf(e1 - mn);
        float rsum = p0 + p1;
#pragma unroll
        for (int msk = 1; msk < 16; msk <<= 1)
          rsum += __shfl_xor(rsum, msk, 32);
        l_i[v] = l_i[v] * corr + rsum;
        m_i[v] = mn;
#pragma unroll
        for (int s = 0; s < 4; ++s) acc[s].f[v] *= corr;
        // stash P (bf16) in per-wave LDS to re-layout C-frag -> A-frag
        ldsP[wv][(v + 8 * hl) * 32 + r]      = f2bf(p0);
        ldsP[wv][(v + 8 * hl) * 32 + 16 + r] = f2bf(p1);
      }
      // O += P @ V
      const v16bf pa = load_a_frag(&ldsP[wv][0], 32);
#pragma unroll
      for (int s = 0; s < 4; ++s) {
        const v16bf vb = load_bt_frag(&ldsV[(s * 16) * 32], 32);
        acc[s].v = WMMA_BF16(pa, vb, acc[s].v);
      }
    }
  }

  // Normalize and write y[b][t][h*64+d] in bf16
  const int b = bh >> 4, hh = bh & 15;
#pragma unroll
  for (int v = 0; v < 8; ++v) {
    const float inv = 1.0f / l_i[v];
    const int t = q0 + v + 8 * hl;
#pragma unroll
    for (int s = 0; s < 4; ++s) {
      const int d = s * 16 + r;
      Y[((size_t)(b * 2048 + t)) * 1024 + hh * 64 + d] = f2bf(acc[s].f[v] * inv);
    }
  }
}

// ---------------------------------------------------------------------------
// Kernel 3: out = y @ w_proj + b_proj  (M=4096, N=1024, K=1024), f32 output.
// ---------------------------------------------------------------------------
__global__ __launch_bounds__(256) void proj_gemm_kernel(
    const __bf16* __restrict__ Yin, const float* __restrict__ w,
    const float* __restrict__ bias, float* __restrict__ out)
{
  __shared__ __attribute__((aligned(16))) __bf16 ldsA[128 * 64];
  __shared__ __attribute__((aligned(16))) __bf16 ldsB[128 * 64];
  const int tid  = threadIdx.x;
  const int m0   = blockIdx.y * 128;
  const int n0   = blockIdx.x * 128;
  const int wv   = tid >> 5;
  const int lane = tid & 31;
  const int r    = lane & 15;
  const int hl   = lane >> 4;

  F8 acc[8];
#pragma unroll
  for (int s = 0; s < 8; ++s)
#pragma unroll
    for (int v = 0; v < 8; ++v) acc[s].f[v] = 0.0f;

  for (int k0 = 0; k0 < 1024; k0 += 64) {
    __syncthreads();
    // Stage A tile (already bf16): straight 16B copies
#pragma unroll
    for (int i = 0; i < 4; ++i) {
      const int e   = (i * 256 + tid) * 8;           // 1024 uint4 total
      const int row = e >> 6, col = e & 63;
      *reinterpret_cast<uint4*>(&ldsA[row * 64 + col]) =
          *reinterpret_cast<const uint4*>(Yin + (size_t)(m0 + row) * 1024 + k0 + col);
    }
    // Stage Bt tile from w_proj fp32
#pragma unroll
    for (int i = 0; i < 8; ++i) {
      const int e  = (i * 256 + tid) * 4;
      const int kk = e >> 7, nn = e & 127;
      const float4 wa =
          *reinterpret_cast<const float4*>(w + (size_t)(k0 + kk) * 1024 + n0 + nn);
      ldsB[(nn + 0) * 64 + kk] = f2bf(wa.x);
      ldsB[(nn + 1) * 64 + kk] = f2bf(wa.y);
      ldsB[(nn + 2) * 64 + kk] = f2bf(wa.z);
      ldsB[(nn + 3) * 64 + kk] = f2bf(wa.w);
    }
    __syncthreads();
#pragma unroll
    for (int kk = 0; kk < 64; kk += 32) {
      const v16bf a = load_a_frag(&ldsA[(wv * 16) * 64 + kk], 64);
#pragma unroll
      for (int s = 0; s < 8; ++s) {
        const v16bf b = load_bt_frag(&ldsB[(s * 16) * 64 + kk], 64);
        acc[s].v = WMMA_BF16(a, b, acc[s].v);
      }
    }
  }

#pragma unroll
  for (int s = 0; s < 8; ++s) {
    const int n    = n0 + s * 16 + r;
    const float bn = bias[n];
#pragma unroll
    for (int v = 0; v < 8; ++v) {
      const int m = m0 + wv * 16 + v + 8 * hl;
      out[(size_t)m * 1024 + n] = acc[s].f[v] + bn;
    }
  }
}

// ---------------------------------------------------------------------------
extern "C" void kernel_launch(void* const* d_in, const int* in_sizes, int n_in,
                              void* d_out, int out_size, void* d_ws, size_t ws_size,
                              hipStream_t stream) {
  (void)in_sizes; (void)n_in; (void)out_size; (void)ws_size;
  const float* x      = (const float*)d_in[0];
  const float* w_attn = (const float*)d_in[1];
  const float* b_attn = (const float*)d_in[2];
  const float* w_proj = (const float*)d_in[3];
  const float* b_proj = (const float*)d_in[4];
  float* out = (float*)d_out;

  // Workspace layout (bf16): Q | K | V^T | Y, each B*H*T*64 = 4,194,304 elems (8 MB)
  const size_t NPH = (size_t)2 * 16 * 2048 * 64;   // = 4194304
  __bf16* Q  = (__bf16*)d_ws;
  __bf16* K  = Q  + NPH;
  __bf16* Vt = K  + NPH;
  __bf16* Y  = Vt + NPH;

  qkv_gemm_kernel  <<<dim3(24, 32), 256, 0, stream>>>(x, w_attn, b_attn, Q, K, Vt);
  flash_attn_kernel<<<dim3(32, 32), 128, 0, stream>>>(Q, K, Vt, Y);
  proj_gemm_kernel <<<dim3( 8, 32), 256, 0, stream>>>(Y, w_proj, b_proj, out);
}
